// MLA_rope_38800734552689
// MI455X (gfx1250) — compile-verified, matrix operand order checked by reading, not run on hardware
//
#include <hip/hip_runtime.h>
#include <math.h>

// ---------------------------------------------------------------------------
// Types for CDNA5 WMMA (gfx1250, wave32)
// ---------------------------------------------------------------------------
typedef __bf16 bf16_t;
typedef __bf16 v16bf __attribute__((ext_vector_type(16)));
typedef __bf16 v8bf  __attribute__((ext_vector_type(8)));
typedef __bf16 v4bf  __attribute__((ext_vector_type(4)));
typedef float  v8f   __attribute__((ext_vector_type(8)));
typedef int    v4i_t __attribute__((ext_vector_type(4)));

#define WMMA_BF16(a, b, c) \
  __builtin_amdgcn_wmma_f32_16x16x32_bf16(false, (a), false, (b), (short)0, (c), false, false)

// CDNA5 async global->LDS copies (ASYNCcnt-tracked), guarded so the file still
// compiles on toolchains without the builtin (falls back to batched reg copy).
#if defined(__has_builtin)
#if __has_builtin(__builtin_amdgcn_global_load_async_to_lds_b128)
#define HAVE_ASYNC_LDS 1
#endif
#endif

#ifdef HAVE_ASYNC_LDS
// Signature (verified by compiler diagnostic): (v4i AS1*, v4i AS3*, imm, imm cpol)
__device__ __forceinline__ void async_copy16(void* lds, const void* g) {
  __builtin_amdgcn_global_load_async_to_lds_b128(
      (__attribute__((address_space(1))) v4i_t*)g,
      (__attribute__((address_space(3))) v4i_t*)lds, 0, 0);
}
#endif

__device__ __forceinline__ void wait_async0() {
#if defined(__has_builtin) && __has_builtin(__builtin_amdgcn_s_wait_asynccnt)
  __builtin_amdgcn_s_wait_asynccnt(0);
#else
  asm volatile("s_wait_asynccnt 0x0" ::: "memory");
#endif
}

// Problem constants (match reference)
constexpr int Bb  = 2;
constexpr int Ss  = 2048;
constexpr int Dd  = 2048;
constexpr int LQn = 1536;
constexpr int LKV = 512;
constexpr int Hh  = 16;
constexpr int Mrows = Bb * Ss;   // 4096 rows for all projections

// ---------------------------------------------------------------------------
// fp32 -> bf16 converter (4-wide)
// ---------------------------------------------------------------------------
__global__ void k_cvt4(const float4* __restrict__ in, v4bf* __restrict__ out, int n4) {
  int i = blockIdx.x * blockDim.x + threadIdx.x;
  if (i >= n4) return;
  float4 f = in[i];
  v4bf o;
  o[0] = (bf16_t)f.x; o[1] = (bf16_t)f.y; o[2] = (bf16_t)f.z; o[3] = (bf16_t)f.w;
  out[i] = o;
}

// ---------------------------------------------------------------------------
// Wo (2048x2048 f32) -> group-16 row-summed Wo_p (128x2048 bf16)
// Exact because v is jnp.repeat'ed 16x before o@Wo.
// ---------------------------------------------------------------------------
__global__ void k_pool_wo(const float* __restrict__ Wo, bf16_t* __restrict__ Wop) {
  int idx = blockIdx.x * blockDim.x + threadIdx.x;
  if (idx >= 128 * 2048) return;
  int r = idx >> 11, n = idx & 2047;
  float s = 0.f;
#pragma unroll
  for (int i = 0; i < 16; i++) s += Wo[(size_t)(r * 16 + i) * 2048 + n];
  Wop[idx] = (bf16_t)s;
}

// ---------------------------------------------------------------------------
// Tiled WMMA GEMM: C(MxN) = A(MxK,bf16) * B(KxN,bf16) + bias, optional residual
// BM=128 BN=64 BK=64, 256 threads = 8 waves, each wave owns a 32x32 C tile.
// Double-buffered LDS: async global->LDS copies of tile n+1 overlap WMMA of n.
// ---------------------------------------------------------------------------
#define GBM 128
#define GBN 64
#define GBK 64
#define APAD 16   // stride 80 elems = 160B: 16B-aligned rows, low bank conflicts

__global__ __launch_bounds__(256) void k_gemm_bf16(
    const bf16_t* __restrict__ A, const bf16_t* __restrict__ Bw,
    const float* __restrict__ bias, const float* __restrict__ residual,
    float* __restrict__ outF, bf16_t* __restrict__ outB,
    int M, int N, int K)
{
  __shared__ __align__(16) bf16_t As[2][GBM][GBK + APAD];
  __shared__ __align__(16) bf16_t Bs[2][GBN][GBK + APAD];   // transposed: [n][k]

  const int tid = threadIdx.x;
  const int lane = tid & 31, w = tid >> 5;
  const int lane16 = lane & 15, hi = lane >> 4;
  const int wm = w & 3, wn = w >> 2;                 // 4 waves in M, 2 in N
  const int bm = blockIdx.y * GBM, bn = blockIdx.x * GBN;
  const int klo = hi ? 8 : 0;                        // ISA 16-bit A/B K pattern

  v8f acc[2][2];
  acc[0][0] = (v8f){}; acc[0][1] = (v8f){}; acc[1][0] = (v8f){}; acc[1][1] = (v8f){};

  // Per-thread staging coordinates
  const int arow[4] = { (tid + 0) >> 3, (tid + 256) >> 3, (tid + 512) >> 3, (tid + 768) >> 3 };
  const int acol = (tid & 7) * 8;
  const int bkrow[2] = { (tid + 0) >> 3, (tid + 256) >> 3 };
  const int bncol = (tid & 7) * 8;

  // Stage tile at column offset kk into LDS buffer `buf`
  auto stage = [&](int buf, int kk) {
#ifdef HAVE_ASYNC_LDS
    // A tile: 128x64 bf16, 4 async b128 copies per thread (no VGPR round-trip)
#pragma unroll
    for (int j = 0; j < 4; j++)
      async_copy16(&As[buf][arow[j]][acol], A + (size_t)(bm + arow[j]) * K + kk + acol);
    // B tile: needs transpose -> register path, loads batched before stores
    v8bf breg[2];
#pragma unroll
    for (int j = 0; j < 2; j++)
      breg[j] = *(const v8bf*)(Bw + (size_t)(kk + bkrow[j]) * N + bn + bncol);
#pragma unroll
    for (int j = 0; j < 2; j++)
#pragma unroll
      for (int i = 0; i < 8; i++) Bs[buf][bncol + i][bkrow[j]] = breg[j][i];
#else
    // Batched: issue ALL global loads before any LDS store (max MLP)
    v8bf areg[4], breg[2];
#pragma unroll
    for (int j = 0; j < 4; j++)
      areg[j] = *(const v8bf*)(A + (size_t)(bm + arow[j]) * K + kk + acol);
#pragma unroll
    for (int j = 0; j < 2; j++)
      breg[j] = *(const v8bf*)(Bw + (size_t)(kk + bkrow[j]) * N + bn + bncol);
#pragma unroll
    for (int j = 0; j < 4; j++)
      *(v8bf*)&As[buf][arow[j]][acol] = areg[j];
#pragma unroll
    for (int j = 0; j < 2; j++)
#pragma unroll
      for (int i = 0; i < 8; i++) Bs[buf][bncol + i][bkrow[j]] = breg[j][i];
#endif
    // L2 prefetch one tile beyond the one being staged
    if (kk + GBK < K) {
      __builtin_prefetch(A + (size_t)(bm + (tid >> 1)) * K + kk + GBK, 0, 3);
      __builtin_prefetch(Bw + (size_t)(kk + GBK + (tid >> 2)) * N + bn, 0, 3);
    }
  };

  const int T = K / GBK;
  stage(0, 0);
  for (int n = 0; n < T; n++) {
    const int cur = n & 1;
    wait_async0();        // tile n's async A copies have landed in LDS
    __syncthreads();      // all waves' stage-n writes visible; frees buf cur^1
    if (n + 1 < T) stage(cur ^ 1, (n + 1) * GBK);   // overlaps compute below

#pragma unroll
    for (int ks = 0; ks < 2; ks++) {
      const int kc = ks * 32 + klo;
      v16bf af[2], bfm[2];
#pragma unroll
      for (int mt = 0; mt < 2; mt++) {
        int row = wm * 32 + mt * 16 + lane16;
        v8bf lo = *(const v8bf*)&As[cur][row][kc];
        v8bf hh = *(const v8bf*)&As[cur][row][kc + 16];
        af[mt] = __builtin_shufflevector(lo, hh, 0,1,2,3,4,5,6,7,8,9,10,11,12,13,14,15);
      }
#pragma unroll
      for (int nt = 0; nt < 2; nt++) {
        int col = wn * 32 + nt * 16 + lane16;
        v8bf lo = *(const v8bf*)&Bs[cur][col][kc];
        v8bf hh = *(const v8bf*)&Bs[cur][col][kc + 16];
        bfm[nt] = __builtin_shufflevector(lo, hh, 0,1,2,3,4,5,6,7,8,9,10,11,12,13,14,15);
      }
#pragma unroll
      for (int mt = 0; mt < 2; mt++)
#pragma unroll
        for (int nt = 0; nt < 2; nt++)
          acc[mt][nt] = WMMA_BF16(af[mt], bfm[nt], acc[mt][nt]);
    }
  }

  // Epilogue: C layout = lane holds col n=lane16, rows m=r+hi*8
#pragma unroll
  for (int mt = 0; mt < 2; mt++) {
#pragma unroll
    for (int nt = 0; nt < 2; nt++) {
      int gn = bn + wn * 32 + nt * 16 + lane16;
      float bv = bias ? bias[gn] : 0.f;
#pragma unroll
      for (int r = 0; r < 8; r++) {
        int gm = bm + wm * 32 + mt * 16 + r + hi * 8;
        size_t o = (size_t)gm * N + gn;
        float c = acc[mt][nt][r] + bv;
        if (outF) outF[o] = c + (residual ? residual[o] : 0.f);
        else      outB[o] = (bf16_t)c;
      }
    }
  }
}

// ---------------------------------------------------------------------------
// Build pooled Q: qpool[b,h,s,0..15] = (1/sqrt(192)) * sum_{i<16} qcat[h*192+16t+i]
// qcat = [q_c (2048) | rope(q_p) (1024, half=512)], t in 0..11, pad 12..15 = 0
// ---------------------------------------------------------------------------
__global__ void k_build_qpool(const bf16_t* __restrict__ qc, const bf16_t* __restrict__ qp,
                              const int* __restrict__ pos, bf16_t* __restrict__ qpool) {
  int idx = blockIdx.x * blockDim.x + threadIdx.x;     // (b*H + h)*S + s
  if (idx >= Bb * Hh * Ss) return;
  int s = idx & (Ss - 1);
  int h = (idx / Ss) & (Hh - 1);
  int b = idx / (Ss * Hh);
  float p = (float)pos[b * Ss + s];
  const bf16_t* qcrow = qc + (size_t)(b * Ss + s) * 2048;
  const bf16_t* qprow = qp + (size_t)(b * Ss + s) * 1024;
  const float scale = 0.07216878364870323f;            // 1/sqrt(192)
  bf16_t* dst = qpool + (size_t)idx * 16;
  for (int t = 0; t < 12; t++) {
    float acc = 0.f;
    for (int i = 0; i < 16; i++) {
      int j = h * 192 + t * 16 + i;
      float val;
      if (j < 2048) {
        val = (float)qcrow[j];
      } else {
        int jj = j - 2048;
        int i0 = (jj < 512) ? jj : jj - 512;
        float ang = p * __expf(-9.210340371976184f * (float)i0 * (1.0f / 512.0f));
        float sn, cs; __sincosf(ang, &sn, &cs);
        float x1 = (float)qprow[i0], x2 = (float)qprow[i0 + 512];
        val = (jj < 512) ? (x1 * cs - x2 * sn) : (x1 * sn + x2 * cs);
      }
      acc += val;
    }
    dst[t] = (bf16_t)(acc * scale);
  }
  for (int t = 12; t < 16; t++) dst[t] = (bf16_t)0.f;
}

// ---------------------------------------------------------------------------
// Build per-head K (12 valid of 16) and V (8 valid of 16), layouts (B,H,S,16)
// ---------------------------------------------------------------------------
__global__ void k_build_kv(const bf16_t* __restrict__ kc, const bf16_t* __restrict__ kr,
                           const bf16_t* __restrict__ vv, const int* __restrict__ pos,
                           bf16_t* __restrict__ kpad, bf16_t* __restrict__ vpad) {
  int idx = blockIdx.x * blockDim.x + threadIdx.x;
  if (idx >= Bb * Hh * Ss) return;
  int s = idx & (Ss - 1);
  int h = (idx / Ss) & (Hh - 1);
  int b = idx / (Ss * Hh);
  float p = (float)pos[b * Ss + s];
  const bf16_t* kcrow = kc + (size_t)(b * Ss + s) * 128;
  const bf16_t* krrow = kr + (size_t)(b * Ss + s) * 64;
  const bf16_t* vrow  = vv + (size_t)(b * Ss + s) * 128;
  bf16_t* kd = kpad + (size_t)idx * 16;
  bf16_t* vd = vpad + (size_t)idx * 16;
  for (int t = 0; t < 12; t++) {
    int j = h * 12 + t;
    float val;
    if (j < 128) {
      val = (float)kcrow[j];
    } else {
      int jj = j - 128;
      int i0 = (jj < 32) ? jj : jj - 32;
      float ang = p * __expf(-9.210340371976184f * (float)i0 * (1.0f / 32.0f));
      float sn, cs; __sincosf(ang, &sn, &cs);
      float x1 = (float)krrow[i0], x2 = (float)krrow[i0 + 32];
      val = (jj < 32) ? (x1 * cs - x2 * sn) : (x1 * sn + x2 * cs);
    }
    kd[t] = (bf16_t)val;
  }
  for (int t = 12; t < 16; t++) kd[t] = (bf16_t)0.f;
  for (int u = 0; u < 8; u++)  vd[u] = vrow[h * 8 + u];
  for (int u = 8; u < 16; u++) vd[u] = (bf16_t)0.f;
}

// ---------------------------------------------------------------------------
// Flash attention on pooled 16-dim heads. Grid (B*H, S/128), 8 waves x 16 rows.
// Double-buffered K/V staging; per 32-key step/wave: 3 WMMAs + online softmax.
// ---------------------------------------------------------------------------
__global__ __launch_bounds__(256) void k_flash(
    const bf16_t* __restrict__ qpool, const bf16_t* __restrict__ kpad,
    const bf16_t* __restrict__ vpad, bf16_t* __restrict__ o8)
{
  __shared__ __align__(16) bf16_t kbuf[2][32][16];
  __shared__ __align__(16) bf16_t vT[2][16][32];       // [dv][key]
  __shared__ __align__(16) bf16_t pbuf[8][16][32];     // per-wave P transpose

  const int bh = blockIdx.x;                 // b*H + h
  const int tile = blockIdx.y;
  const int b = bh >> 4, h = bh & 15;
  const int tid = threadIdx.x, lane = tid & 31, w = tid >> 5;
  const int lane16 = lane & 15, hi = lane >> 4;
  const int klo = hi ? 8 : 0;

  const v8bf z8 = (v8bf){};

  // Q A-fragment: 16 valid dims, upper K 16..31 zero-padded
  const int qrowA = tile * 128 + w * 16 + lane16;
  v8bf qlo = *(const v8bf*)(qpool + ((size_t)bh * Ss + qrowA) * 16 + klo);
  v16bf aq = __builtin_shufflevector(qlo, z8, 0,1,2,3,4,5,6,7,8,9,10,11,12,13,14,15);

  float mrow[8], lrow[8];
#pragma unroll
  for (int r = 0; r < 8; r++) { mrow[r] = -INFINITY; lrow[r] = 0.f; }
  v8f oacc = (v8f){};

  auto stageKV = [&](int buf, int kb) {
    if (tid < 64) {                    // waves 0-1: stage K rows
      int key = tid >> 1, half = (tid & 1) * 8;
#ifdef HAVE_ASYNC_LDS
      async_copy16(&kbuf[buf][key][half],
                   kpad + ((size_t)bh * Ss + kb + key) * 16 + half);
#else
      *(v8bf*)&kbuf[buf][key][half] =
          *(const v8bf*)(kpad + ((size_t)bh * Ss + kb + key) * 16 + half);
#endif
    } else if (tid < 128) {            // waves 2-3: stage V transposed
      int t = tid - 64;
      int key = t >> 1, half = (t & 1) * 8;
      v8bf d = *(const v8bf*)(vpad + ((size_t)bh * Ss + kb + key) * 16 + half);
#pragma unroll
      for (int i = 0; i < 8; i++) vT[buf][half + i][key] = d[i];
    }
  };

  const int nIter = (tile + 1) * 4;          // 32-key steps (causal bound)
  stageKV(0, 0);
  for (int n = 0; n < nIter; n++) {
    const int cur = n & 1;
    const int kb = n * 32;
    wait_async0();
    __syncthreads();
    if (n + 1 < nIter) stageKV(cur ^ 1, kb + 32);   // overlaps compute below

    // Scores: two 16x16 tiles over 32 keys
    float sc[2][8];
#pragma unroll
    for (int nt = 0; nt < 2; nt++) {
      int nn = nt * 16 + lane16;
      v8bf blo = *(const v8bf*)&kbuf[cur][nn][klo];
      v16bf bk = __builtin_shufflevector(blo, z8, 0,1,2,3,4,5,6,7,8,9,10,11,12,13,14,15);
      v8f s = (v8f){};
      s = WMMA_BF16(aq, bk, s);
      int gkey = kb + nn;
#pragma unroll
      for (int r = 0; r < 8; r++) {
        int qr = tile * 128 + w * 16 + r + hi * 8;
        sc[nt][r] = (gkey > qr) ? -1e30f : s[r];
      }
    }

    // Online softmax; rows live within a 16-lane half -> shfl_xor 1,2,4,8
#pragma unroll
    for (int r = 0; r < 8; r++) {
      float mx = fmaxf(sc[0][r], sc[1][r]);
      for (int msk = 1; msk < 16; msk <<= 1) mx = fmaxf(mx, __shfl_xor(mx, msk, 32));
      float mnew = fmaxf(mrow[r], mx);
      float rescale = __expf(mrow[r] - mnew);
      float p0 = __expf(sc[0][r] - mnew);
      float p1 = __expf(sc[1][r] - mnew);
      float ps = p0 + p1;
      for (int msk = 1; msk < 16; msk <<= 1) ps += __shfl_xor(ps, msk, 32);
      lrow[r] = lrow[r] * rescale + ps;
      mrow[r] = mnew;
      oacc[r] *= rescale;
      int m_ = r + hi * 8;
      pbuf[w][m_][lane16]      = (bf16_t)p0;
      pbuf[w][m_][16 + lane16] = (bf16_t)p1;
    }
    asm volatile("s_wait_dscnt 0x0" ::: "memory");   // wave-local LDS RAW

    // P A-fragment (16x32 keys) and V B-fragment (32 keys x 16 dv)
    v8bf plo = *(const v8bf*)&pbuf[w][lane16][klo];
    v8bf phi = *(const v8bf*)&pbuf[w][lane16][klo + 16];
    v16bf pf = __builtin_shufflevector(plo, phi, 0,1,2,3,4,5,6,7,8,9,10,11,12,13,14,15);
    v8bf vlo = *(const v8bf*)&vT[cur][lane16][klo];
    v8bf vhi = *(const v8bf*)&vT[cur][lane16][klo + 16];
    v16bf vf = __builtin_shufflevector(vlo, vhi, 0,1,2,3,4,5,6,7,8,9,10,11,12,13,14,15);
    oacc = WMMA_BF16(pf, vf, oacc);
  }

  // Emit o8 (B,S,H*8) bf16; valid dv cols are 0..7
  if (lane16 < 8) {
#pragma unroll
    for (int r = 0; r < 8; r++) {
      int qr = tile * 128 + w * 16 + r + hi * 8;
      float v = oacc[r] / lrow[r];
      o8[((size_t)b * Ss + qr) * 128 + h * 8 + lane16] = (bf16_t)v;
    }
  }
}

// ---------------------------------------------------------------------------
// Host-side launch
// ---------------------------------------------------------------------------
extern "C" void kernel_launch(void* const* d_in, const int* in_sizes, int n_in,
                              void* d_out, int out_size, void* d_ws, size_t ws_size,
                              hipStream_t stream)
{
  (void)in_sizes; (void)n_in; (void)out_size; (void)ws_size;

  const float* x     = (const float*)d_in[0];
  const float* Wq_d  = (const float*)d_in[1];
  const float* bq_d  = (const float*)d_in[2];
  const float* Wkv_d = (const float*)d_in[3];
  const float* bkv_d = (const float*)d_in[4];
  const float* Wq_uc = (const float*)d_in[5];
  const float* bq_uc = (const float*)d_in[6];
  const float* Wq_up = (const float*)d_in[7];
  const float* bq_up = (const float*)d_in[8];
  const float* Wk_uc = (const float*)d_in[9];
  const float* bk_uc = (const float*)d_in[10];
  const float* Wk_up = (const float*)d_in[11];
  const float* bk_up = (const float*)d_in[12];
  const float* Wv_u  = (const float*)d_in[13];
  const float* bv_u  = (const float*)d_in[14];
  const float* Wo    = (const float*)d_in[15];
  const float* bo    = (const float*)d_in[16];
  const int*   pos   = (const int*)d_in[17];
  float* out = (float*)d_out;

  char* ws = (char*)d_ws;
  size_t off = 0;
  auto alloc = [&](size_t bytes) -> char* {
    char* p = ws + off;
    off += (bytes + 255) & ~(size_t)255;
    return p;
  };

  bf16_t* xb     = (bf16_t*)alloc((size_t)Mrows * Dd * 2);
  bf16_t* Wqd_b  = (bf16_t*)alloc((size_t)Dd * LQn * 2);
  bf16_t* Wkvd_b = (bf16_t*)alloc((size_t)Dd * LKV * 2);
  bf16_t* Wquc_b = (bf16_t*)alloc((size_t)LQn * 2048 * 2);
  bf16_t* Wqup_b = (bf16_t*)alloc((size_t)LQn * 1024 * 2);
  bf16_t* Wkuc_b = (bf16_t*)alloc((size_t)LKV * 128 * 2);
  bf16_t* Wkup_b = (bf16_t*)alloc((size_t)LKV * 64 * 2);
  bf16_t* Wvu_b  = (bf16_t*)alloc((size_t)LKV * 128 * 2);
  bf16_t* Wop_b  = (bf16_t*)alloc((size_t)128 * 2048 * 2);
  bf16_t* qlat_b = (bf16_t*)alloc((size_t)Mrows * LQn * 2);
  bf16_t* kvlat_b= (bf16_t*)alloc((size_t)Mrows * LKV * 2);
  bf16_t* qc_b   = (bf16_t*)alloc((size_t)Mrows * 2048 * 2);
  bf16_t* qp_b   = (bf16_t*)alloc((size_t)Mrows * 1024 * 2);
  bf16_t* kc_b   = (bf16_t*)alloc((size_t)Mrows * 128 * 2);
  bf16_t* kr_b   = (bf16_t*)alloc((size_t)Mrows * 64 * 2);
  bf16_t* v_b    = (bf16_t*)alloc((size_t)Mrows * 128 * 2);
  bf16_t* qpool_b= (bf16_t*)alloc((size_t)Bb * Hh * Ss * 16 * 2);
  bf16_t* kpad_b = (bf16_t*)alloc((size_t)Bb * Hh * Ss * 16 * 2);
  bf16_t* vpad_b = (bf16_t*)alloc((size_t)Bb * Hh * Ss * 16 * 2);
  bf16_t* o8_b   = (bf16_t*)alloc((size_t)Mrows * 128 * 2);

  auto cvt = [&](const float* src, bf16_t* dst, size_t n) {
    int n4 = (int)(n / 4);
    k_cvt4<<<(n4 + 255) / 256, 256, 0, stream>>>((const float4*)src, (v4bf*)dst, n4);
  };
  cvt(x,     xb,     (size_t)Mrows * Dd);
  cvt(Wq_d,  Wqd_b,  (size_t)Dd * LQn);
  cvt(Wkv_d, Wkvd_b, (size_t)Dd * LKV);
  cvt(Wq_uc, Wquc_b, (size_t)LQn * 2048);
  cvt(Wq_up, Wqup_b, (size_t)LQn * 1024);
  cvt(Wk_uc, Wkuc_b, (size_t)LKV * 128);
  cvt(Wk_up, Wkup_b, (size_t)LKV * 64);
  cvt(Wv_u,  Wvu_b,  (size_t)LKV * 128);

  k_pool_wo<<<(128 * 2048) / 256, 256, 0, stream>>>(Wo, Wop_b);

  auto gemm = [&](const bf16_t* A, const bf16_t* Bw, const float* bias,
                  const float* resid, float* oF, bf16_t* oB, int N, int K) {
    dim3 g(N / GBN, Mrows / GBM);
    k_gemm_bf16<<<g, 256, 0, stream>>>(A, Bw, bias, resid, oF, oB, Mrows, N, K);
  };

  // Down projections
  gemm(xb, Wqd_b,  bq_d,  nullptr, nullptr, qlat_b,  LQn, Dd);
  gemm(xb, Wkvd_b, bkv_d, nullptr, nullptr, kvlat_b, LKV, Dd);
  // Up projections
  gemm(qlat_b,  Wquc_b, bq_uc, nullptr, nullptr, qc_b, 2048, LQn);
  gemm(qlat_b,  Wqup_b, bq_up, nullptr, nullptr, qp_b, 1024, LQn);
  gemm(kvlat_b, Wkuc_b, bk_uc, nullptr, nullptr, kc_b, 128, LKV);
  gemm(kvlat_b, Wkup_b, bk_up, nullptr, nullptr, kr_b, 64,  LKV);
  gemm(kvlat_b, Wvu_b,  bv_u,  nullptr, nullptr, v_b,  128, LKV);

  // RoPE + group-16 pooling
  int nbhs = Bb * Hh * Ss;
  k_build_qpool<<<nbhs / 256, 256, 0, stream>>>(qc_b, qp_b, pos, qpool_b);
  k_build_kv<<<nbhs / 256, 256, 0, stream>>>(kc_b, kr_b, v_b, pos, kpad_b, vpad_b);

  // Attention
  dim3 fg(Bb * Hh, Ss / 128);
  k_flash<<<fg, 256, 0, stream>>>(qpool_b, kpad_b, vpad_b, o8_b);

  // Output projection + bias + residual (fp32 out)
  gemm(o8_b, Wop_b, bo, x, out, nullptr, 2048, 128);
}
